// DeformableTransformerDecoderLayer_55319178772979
// MI455X (gfx1250) — matrix operand order repaired and checked by reading, not compiled, not run on hardware
//
#include <hip/hip_runtime.h>
#include <math.h>

// ---------------------------------------------------------------- types ----
typedef __bf16 bf16;
typedef bf16  bf16x16 __attribute__((ext_vector_type(16)));
typedef bf16  bf16x8  __attribute__((ext_vector_type(8)));
typedef float v8f     __attribute__((ext_vector_type(8)));

__device__ __forceinline__ bf16 f2bf(float f) {
    unsigned u = __float_as_uint(f);
    unsigned r = (u + 0x7FFFu + ((u >> 16) & 1u)) >> 16;   // round-to-nearest-even
    unsigned short s = (unsigned short)r;
    return __builtin_bit_cast(bf16, s);
}
__device__ __forceinline__ float bf2f(bf16 b) {
    unsigned short s = __builtin_bit_cast(unsigned short, b);
    return __uint_as_float(((unsigned)s) << 16);
}

// ------------------------------------------------------------- constants ---
#define NB   8
#define LQC  900
#define CCH  256
#define NHH  8
#define NLL  4
#define NPP  4
#define DH   32
#define LENV 21760
#define M1   (NB * LQC)      // 7200 rows
#define M2   (NB * LENV)     // 174080 rows
#define QT   57              // ceil(900/16)

// ------------------------------------------------- elementwise converts ----
__global__ void k_f32_to_bf16(const float* __restrict__ in, bf16* __restrict__ out, size_t n) {
    size_t i = (size_t)blockIdx.x * blockDim.x + threadIdx.x;
    if (i < n) out[i] = f2bf(in[i]);
}
__global__ void k_add_to_bf16(const float* __restrict__ a, const float* __restrict__ b,
                              bf16* __restrict__ out, size_t n) {
    size_t i = (size_t)blockIdx.x * blockDim.x + threadIdx.x;
    if (i < n) out[i] = f2bf(a[i] + b[i]);
}
__global__ void k_mask_zero(bf16* __restrict__ value, const unsigned char* __restrict__ mask, int rows) {
    int r = blockIdx.x * blockDim.x + threadIdx.x;
    if (r >= rows || !mask[r]) return;
    bf16 z = f2bf(0.f);
    bf16* p = value + (size_t)r * CCH;
    for (int c = 0; c < CCH; ++c) p[c] = z;
}

// --------------------------------------------------------- WMMA GEMM ------
// out[M, Nd] = act(A[M,K] @ W[Nd,K]^T + bias), bf16 operands, f32 accum.
// Block = 8 waves as 2(M) x 4(N); each wave computes a 16x64 tile with 4
// accumulators, reusing its A fragment across 4 v_wmma_f32_16x16x32_bf16
// per K-step. All 4 B fragments are fetched into distinct registers before
// the WMMA chain so loads overlap the matrix pipe (partial loadcnt waits).
// Requires M % 32 == 0, Nd % 64 == 0.
__global__ __launch_bounds__(256) void k_gemm_bf16(
    const bf16* __restrict__ A, const bf16* __restrict__ W,
    const float* __restrict__ bias, int M, int Nd, int K,
    float* __restrict__ outF, bf16* __restrict__ outB, int relu)
{
    int wave  = threadIdx.x >> 5;
    int lane  = threadIdx.x & 31;
    int rtile = wave >> 2;                  // 0..1
    int ctile = wave & 3;                   // 0..3
    int row0  = blockIdx.x * 32 + rtile * 16;
    int col0  = blockIdx.y * 256 + ctile * 64;
    if (col0 >= Nd) return;                 // wave-uniform, EXEC stays full

    int l16  = lane & 15;
    int half = lane >> 4;
    int kb   = half * 8;
    size_t arow = (size_t)(row0 + l16) * K;
    size_t wrow[4];
    for (int c = 0; c < 4; ++c) wrow[c] = (size_t)(col0 + c * 16 + l16) * K;

    v8f acc[4];
    for (int c = 0; c < 4; ++c)
        for (int i = 0; i < 8; ++i) acc[c][i] = 0.f;

    for (int kk = 0; kk < K; kk += 32) {
        __builtin_prefetch(A + arow + kk + 256, 0, 1);   // global_prefetch_b8
        // A fragment: row = row0+l16; K elems {half*16..half*16+15}
        bf16x16 af;
        {
            bf16x8 lo = *(const bf16x8*)(A + arow + kk + half * 16);
            bf16x8 hi = *(const bf16x8*)(A + arow + kk + half * 16 + 8);
            for (int i = 0; i < 8; ++i) { af[i] = lo[i]; af[8 + i] = hi[i]; }
        }
        // all 4 B fragments first (distinct regs) so loads pipeline vs WMMA
        bf16x16 bfr[4];
        for (int c = 0; c < 4; ++c) {
            bf16x8 lo = *(const bf16x8*)(W + wrow[c] + kk + half * 16);
            bf16x8 hi = *(const bf16x8*)(W + wrow[c] + kk + half * 16 + 8);
            for (int i = 0; i < 8; ++i) { bfr[c][i] = lo[i]; bfr[c][8 + i] = hi[i]; }
        }
        for (int c = 0; c < 4; ++c)
            acc[c] = __builtin_amdgcn_wmma_f32_16x16x32_bf16(false, af, false, bfr[c],
                                                             (short)0, acc[c], false, false);
    }
    // D layout: lane -> out col = base+l16 ; VGPR r -> row = row0 + half*8 + r
    for (int c = 0; c < 4; ++c) {
        int   ocol = col0 + c * 16 + l16;
        float bv   = bias ? bias[ocol] : 0.f;
        for (int r = 0; r < 8; ++r) {
            int orow = row0 + half * 8 + r;
            float v = acc[c][r] + bv;
            if (relu) v = fmaxf(v, 0.f);
            size_t idx = (size_t)orow * Nd + ocol;
            if (outF) outF[idx] = v;
            if (outB) outB[idx] = f2bf(v);
        }
    }
}

// ----------------------------------------------- flash attention (MHA) -----
// One wave per (batch, head, 16-query tile). dh = 32 = WMMA K dimension.
__global__ __launch_bounds__(32) void k_flash_attn(
    const bf16* __restrict__ Q, const bf16* __restrict__ Kp, const bf16* __restrict__ Vp,
    float* __restrict__ outF, bf16* __restrict__ outB)
{
    __shared__ bf16 Pt[16 * 32];
    const float scale = 0.17677669529663687f;   // 1/sqrt(32)

    int lane = threadIdx.x & 31;
    int bid  = blockIdx.x;
    int qt   = bid % QT;
    int h    = (bid / QT) % NHH;
    int n    = bid / (QT * NHH);
    int q0   = qt * 16;
    int l16  = lane & 15;
    int half = lane >> 4;
    int kb   = half * 8;

    // Q fragment (A layout), loaded once
    bf16x16 afq;
    {
        const bf16* qrow = Q + ((size_t)(n * LQC) + q0 + l16) * CCH + h * DH;
        bf16x8 lo = *(const bf16x8*)(qrow + half * 16);
        bf16x8 hi = *(const bf16x8*)(qrow + half * 16 + 8);
        for (int i = 0; i < 8; ++i) { afq[i] = lo[i]; afq[8 + i] = hi[i]; }
    }

    float m_[8], l_[8];
    v8f o0, o1;
    for (int r = 0; r < 8; ++r) { m_[r] = -1e30f; l_[r] = 0.f; o0[r] = 0.f; o1[r] = 0.f; }

    for (int kc = 0; kc < LQC; kc += 32) {
        // ---- logits: Q (16x32) @ K^T (32 x 32keys) via two WMMAs ----
        v8f t0, t1;
        for (int r = 0; r < 8; ++r) { t0[r] = 0.f; t1[r] = 0.f; }
        {
            bf16x16 bk;
            const bf16* krow = Kp + ((size_t)(n * LQC) + kc + l16) * CCH + h * DH + half * 16;
            bf16x8 lo = *(const bf16x8*)krow;
            bf16x8 hi = *(const bf16x8*)(krow + 8);
            for (int i = 0; i < 8; ++i) { bk[i] = lo[i]; bk[8 + i] = hi[i]; }
            t0 = __builtin_amdgcn_wmma_f32_16x16x32_bf16(false, afq, false, bk, (short)0, t0, false, false);
        }
        {
            bf16x16 bk;
            const bf16* krow = Kp + ((size_t)(n * LQC) + kc + 16 + l16) * CCH + h * DH + half * 16;
            bf16x8 lo = *(const bf16x8*)krow;
            bf16x8 hi = *(const bf16x8*)(krow + 8);
            for (int i = 0; i < 8; ++i) { bk[i] = lo[i]; bk[8 + i] = hi[i]; }
            t1 = __builtin_amdgcn_wmma_f32_16x16x32_bf16(false, afq, false, bk, (short)0, t1, false, false);
        }
        // scale + key mask
        bool bad0 = (kc + l16)      >= LQC;
        bool bad1 = (kc + 16 + l16) >= LQC;
        for (int r = 0; r < 8; ++r) {
            t0[r] = bad0 ? -1e30f : t0[r] * scale;
            t1[r] = bad1 ? -1e30f : t1[r] * scale;
        }
        // online softmax (row stats shared across 16 lanes of each half)
        float mx[8];
        for (int r = 0; r < 8; ++r) mx[r] = fmaxf(t0[r], t1[r]);
        for (int d = 1; d < 16; d <<= 1)
            for (int r = 0; r < 8; ++r) mx[r] = fmaxf(mx[r], __shfl_xor(mx[r], d, 16));
        float p0s[8], p1s[8], ls[8];
        for (int r = 0; r < 8; ++r) {
            float mn = fmaxf(m_[r], mx[r]);
            float sc = __expf(m_[r] - mn);
            m_[r] = mn;
            p0s[r] = __expf(t0[r] - mn);
            p1s[r] = __expf(t1[r] - mn);
            ls[r]  = p0s[r] + p1s[r];
            l_[r] *= sc; o0[r] *= sc; o1[r] *= sc;
        }
        for (int d = 1; d < 16; d <<= 1)
            for (int r = 0; r < 8; ++r) ls[r] += __shfl_xor(ls[r], d, 16);
        for (int r = 0; r < 8; ++r) l_[r] += ls[r];

        // stage P through LDS to re-fragment into A layout
        for (int r = 0; r < 8; ++r) {
            Pt[(half * 8 + r) * 32 + l16]      = f2bf(p0s[r]);
            Pt[(half * 8 + r) * 32 + 16 + l16] = f2bf(p1s[r]);
        }
        __syncthreads();
        bf16x16 ap;
        for (int i = 0; i < 8; ++i) {
            ap[i]     = Pt[l16 * 32 + half * 16 + i];
            ap[8 + i] = Pt[l16 * 32 + half * 16 + 8 + i];
        }
        // V fragments (B layout): col = dh; K elems = keys half*16 + 0..15
        bf16x16 bv0, bv1;
        {
            const bf16* vbase = Vp + ((size_t)(n * LQC) + kc + half * 16) * CCH + h * DH;
            for (int j = 0; j < 16; ++j) {
                bv0[j] = vbase[(size_t)j * CCH + l16];
                bv1[j] = vbase[(size_t)j * CCH + 16 + l16];
            }
        }
        o0 = __builtin_amdgcn_wmma_f32_16x16x32_bf16(false, ap, false, bv0, (short)0, o0, false, false);
        o1 = __builtin_amdgcn_wmma_f32_16x16x32_bf16(false, ap, false, bv1, (short)0, o1, false, false);
        __syncthreads();
    }

    for (int r = 0; r < 8; ++r) {
        int q = q0 + half * 8 + r;
        if (q < LQC) {
            float inv = 1.0f / l_[r];
            size_t base = ((size_t)(n * LQC) + q) * CCH + h * DH;
            float y0 = o0[r] * inv, y1 = o1[r] * inv;
            outF[base + l16]      = y0;
            outF[base + 16 + l16] = y1;
            outB[base + l16]      = f2bf(y0);
            outB[base + 16 + l16] = f2bf(y1);
        }
    }
}

// ----------------------------------------------- residual + LayerNorm ------
// One wave per row of C=256. outB (optional) = bf16(y + extra) for next GEMM.
__global__ __launch_bounds__(256) void k_add_ln(
    const float* __restrict__ x, const float* __restrict__ res,
    const float* __restrict__ g, const float* __restrict__ bta,
    float* __restrict__ outF, bf16* __restrict__ outB,
    const float* __restrict__ extra, int M)
{
    int row  = blockIdx.x * (blockDim.x >> 5) + (threadIdx.x >> 5);
    if (row >= M) return;
    int lane = threadIdx.x & 31;
    size_t base = (size_t)row * CCH;

    float v[8], s = 0.f;
    for (int i = 0; i < 8; ++i) {
        int c = lane * 8 + i;
        v[i] = x[base + c] + (res ? res[base + c] : 0.f);
        s += v[i];
    }
    for (int d = 1; d < 32; d <<= 1) s += __shfl_xor(s, d, 32);
    float mean = s * (1.f / CCH);
    float vs = 0.f;
    for (int i = 0; i < 8; ++i) { float dv = v[i] - mean; vs += dv * dv; }
    for (int d = 1; d < 32; d <<= 1) vs += __shfl_xor(vs, d, 32);
    float inv = rsqrtf(vs * (1.f / CCH) + 1e-5f);
    for (int i = 0; i < 8; ++i) {
        int c = lane * 8 + i;
        float y = (v[i] - mean) * inv * g[c] + bta[c];
        if (outF) outF[base + c] = y;
        if (outB) outB[base + c] = f2bf(y + (extra ? extra[base + c] : 0.f));
    }
}

// ------------------------------------- softmax over NL*NP=16 per head ------
__global__ void k_softmax16(float* __restrict__ a, int groups) {
    int gidx = blockIdx.x * blockDim.x + threadIdx.x;
    if (gidx >= groups) return;
    float* p = a + (size_t)gidx * 16;
    float m = -1e30f;
    for (int i = 0; i < 16; ++i) m = fmaxf(m, p[i]);
    float s = 0.f;
    for (int i = 0; i < 16; ++i) { float e = __expf(p[i] - m); p[i] = e; s += e; }
    float inv = 1.f / s;
    for (int i = 0; i < 16; ++i) p[i] *= inv;
}

// ------------------------------------------- deformable sampling gather ----
__device__ __forceinline__ float sampleV(const bf16* vl, int xi, int yi, int Ws, int Hs) {
    if (xi < 0 || xi >= Ws || yi < 0 || yi >= Hs) return 0.f;
    return bf2f(vl[((size_t)yi * Ws + xi) * CCH]);
}

__global__ __launch_bounds__(256) void k_msdeform(
    const bf16* __restrict__ value, const float* __restrict__ ref,
    const float* __restrict__ off, const float* __restrict__ aw,
    float* __restrict__ outF, bf16* __restrict__ outB)
{
    const int   Hs_[NLL] = {128, 64, 32, 16};
    const int   Ws_[NLL] = {128, 64, 32, 16};
    const int   st_[NLL] = {0, 16384, 20480, 21504};

    int wid = blockIdx.x * (blockDim.x >> 5) + (threadIdx.x >> 5);
    if (wid >= NB * LQC * NHH) return;
    int lane = threadIdx.x & 31;                 // channel within head
    int h = wid % NHH;
    int t = wid / NHH;
    int q = t % LQC;
    int n = t / LQC;

    size_t rowq = (size_t)n * LQC + q;
    const float* refp = ref + rowq * NLL * 2;
    const float* offp = off + rowq * CCH + h * NLL * NPP * 2;
    const float* awp  = aw  + rowq * 128 + h * NLL * NPP;

    float acc = 0.f;
    for (int l = 0; l < NLL; ++l) {
        float rx = refp[l * 2], ry = refp[l * 2 + 1];
        int Hs = Hs_[l], Ws = Ws_[l];
        const bf16* vl = value + ((size_t)n * LENV + st_[l]) * CCH + h * DH + lane;
        for (int p = 0; p < NPP; ++p) {
            float ox = offp[(l * NPP + p) * 2];
            float oy = offp[(l * NPP + p) * 2 + 1];
            float a  = awp[l * NPP + p];
            float x = rx * Ws + ox - 0.5f;       // (rx + ox/Ws)*Ws - 0.5
            float y = ry * Hs + oy - 0.5f;
            float x0f = floorf(x), y0f = floorf(y);
            int x0 = (int)x0f, y0 = (int)y0f;
            float wx = x - x0f, wy = y - y0f;
            float v00 = sampleV(vl, x0,     y0,     Ws, Hs);
            float v10 = sampleV(vl, x0 + 1, y0,     Ws, Hs);
            float v01 = sampleV(vl, x0,     y0 + 1, Ws, Hs);
            float v11 = sampleV(vl, x0 + 1, y0 + 1, Ws, Hs);
            acc += a * ((1.f - wx) * (1.f - wy) * v00 + wx * (1.f - wy) * v10 +
                        (1.f - wx) * wy * v01 + wx * wy * v11);
        }
    }
    size_t base = rowq * CCH + h * DH + lane;
    outF[base] = acc;
    outB[base] = f2bf(acc);
}

// ---------------------------------------------------------------- host -----
static void launch_gemm(const bf16* A, const bf16* W, const float* bias,
                        int M, int Nd, int K, float* outF, bf16* outB, int relu,
                        hipStream_t stream) {
    dim3 grid(M / 32, (Nd + 255) / 256);
    k_gemm_bf16<<<grid, 256, 0, stream>>>(A, W, bias, M, Nd, K, outF, outB, relu);
}
static void launch_cvt(const float* in, bf16* out, size_t n, hipStream_t stream) {
    k_f32_to_bf16<<<(unsigned)((n + 255) / 256), 256, 0, stream>>>(in, out, n);
}

extern "C" void kernel_launch(void* const* d_in, const int* in_sizes, int n_in,
                              void* d_out, int out_size, void* d_ws, size_t ws_size,
                              hipStream_t stream) {
    (void)in_sizes; (void)n_in; (void)out_size; (void)ws_size;

    const float* tgt      = (const float*)d_in[0];
    const float* qpos     = (const float*)d_in[1];
    const float* refpts   = (const float*)d_in[2];
    const float* src      = (const float*)d_in[3];
    const unsigned char* pmask = (const unsigned char*)d_in[6];
    const float* sa_in_w  = (const float*)d_in[7];
    const float* sa_in_b  = (const float*)d_in[8];
    const float* sa_out_w = (const float*)d_in[9];
    const float* sa_out_b = (const float*)d_in[10];
    const float* off_w    = (const float*)d_in[11];
    const float* off_b    = (const float*)d_in[12];
    const float* aw_w     = (const float*)d_in[13];
    const float* aw_b     = (const float*)d_in[14];
    const float* val_w    = (const float*)d_in[15];
    const float* val_b    = (const float*)d_in[16];
    const float* cao_w    = (const float*)d_in[17];
    const float* cao_b    = (const float*)d_in[18];
    const float* n1_g     = (const float*)d_in[19];
    const float* n1_b     = (const float*)d_in[20];
    const float* n2_g     = (const float*)d_in[21];
    const float* n2_b     = (const float*)d_in[22];
    const float* n3_g     = (const float*)d_in[23];
    const float* n3_b     = (const float*)d_in[24];
    const float* f1_w     = (const float*)d_in[25];
    const float* f1_b     = (const float*)d_in[26];
    const float* f2_w     = (const float*)d_in[27];
    const float* f2_b     = (const float*)d_in[28];

    // bump allocator over d_ws; 32KB granularity gives alignment + tail slack
    char*  base = (char*)d_ws;
    size_t cur  = 0;
    auto alloc = [&](size_t bytes) -> void* {
        void* p = base + cur;
        cur += (bytes + 0x8000 + 0x7FFF) & ~(size_t)0x7FFF;
        return p;
    };

    bf16*  qadd_bf  = (bf16*) alloc((size_t)M1 * CCH * 2);
    bf16*  tgt_bf   = (bf16*) alloc((size_t)M1 * CCH * 2);
    bf16*  wslot    = (bf16*) alloc((size_t)1024 * 256 * 2);   // largest weight
    bf16*  qp_bf    = (bf16*) alloc((size_t)M1 * CCH * 2);
    bf16*  kp_bf    = (bf16*) alloc((size_t)M1 * CCH * 2);
    bf16*  vp_bf    = (bf16*) alloc((size_t)M1 * CCH * 2);
    float* attnF    = (float*)alloc((size_t)M1 * CCH * 4);
    bf16*  attnB    = (bf16*) alloc((size_t)M1 * CCH * 2);
    float* tmpF     = (float*)alloc((size_t)M1 * CCH * 4);     // GEMM f32 outs
    float* tgt1F    = (float*)alloc((size_t)M1 * CCH * 4);     // after LN(n2)
    bf16*  q2_bf    = (bf16*) alloc((size_t)M1 * CCH * 2);     // bf16(tgt1+pos)
    bf16*  src_bf   = (bf16*) alloc((size_t)M2 * CCH * 2);
    bf16*  value_bf = (bf16*) alloc((size_t)M2 * CCH * 2);
    float* offF     = (float*)alloc((size_t)M1 * 256 * 4);
    float* awF      = (float*)alloc((size_t)M1 * 128 * 4);
    float* defF     = (float*)alloc((size_t)M1 * CCH * 4);
    bf16*  defB     = (bf16*) alloc((size_t)M1 * CCH * 2);
    float* tgt3F    = (float*)alloc((size_t)M1 * CCH * 4);     // after LN(n1)
    bf16*  tgt3B    = (bf16*) alloc((size_t)M1 * CCH * 2);
    bf16*  ffnB     = (bf16*) alloc((size_t)M1 * 1024 * 2);
    float* ffnF     = (float*)alloc((size_t)M1 * CCH * 4);

    size_t nAct = (size_t)M1 * CCH;

    // ---- self attention ----
    k_add_to_bf16<<<(unsigned)((nAct + 255) / 256), 256, 0, stream>>>(tgt, qpos, qadd_bf, nAct);
    launch_cvt(tgt, tgt_bf, nAct, stream);
    launch_cvt(sa_in_w, wslot, (size_t)768 * 256, stream);
    launch_gemm(qadd_bf, wslot,             sa_in_b,       M1, 256, 256, nullptr, qp_bf, 0, stream);
    launch_gemm(qadd_bf, wslot + 256 * 256, sa_in_b + 256, M1, 256, 256, nullptr, kp_bf, 0, stream);
    launch_gemm(tgt_bf,  wslot + 512 * 256, sa_in_b + 512, M1, 256, 256, nullptr, vp_bf, 0, stream);
    k_flash_attn<<<NB * NHH * QT, 32, 0, stream>>>(qp_bf, kp_bf, vp_bf, attnF, attnB);
    launch_cvt(sa_out_w, wslot, (size_t)256 * 256, stream);
    launch_gemm(attnB, wslot, sa_out_b, M1, 256, 256, tmpF, nullptr, 0, stream);
    k_add_ln<<<(M1 + 7) / 8, 256, 0, stream>>>(tmpF, tgt, n2_g, n2_b, tgt1F, q2_bf, qpos, M1);

    // ---- deformable cross attention ----
    launch_cvt(src, src_bf, (size_t)M2 * CCH, stream);
    launch_cvt(val_w, wslot, (size_t)256 * 256, stream);
    launch_gemm(src_bf, wslot, val_b, M2, 256, 256, nullptr, value_bf, 0, stream);
    k_mask_zero<<<(M2 + 255) / 256, 256, 0, stream>>>(value_bf, pmask, M2);
    launch_cvt(off_w, wslot, (size_t)256 * 256, stream);
    launch_gemm(q2_bf, wslot, off_b, M1, 256, 256, offF, nullptr, 0, stream);
    launch_cvt(aw_w, wslot, (size_t)128 * 256, stream);
    launch_gemm(q2_bf, wslot, aw_b, M1, 128, 256, awF, nullptr, 0, stream);
    k_softmax16<<<(NB * LQC * NHH + 255) / 256, 256, 0, stream>>>(awF, NB * LQC * NHH);
    k_msdeform<<<(NB * LQC * NHH + 7) / 8, 256, 0, stream>>>(value_bf, refpts, offF, awF, defF, defB);
    launch_cvt(cao_w, wslot, (size_t)256 * 256, stream);
    launch_gemm(defB, wslot, cao_b, M1, 256, 256, tmpF, nullptr, 0, stream);
    k_add_ln<<<(M1 + 7) / 8, 256, 0, stream>>>(tmpF, tgt1F, n1_g, n1_b, tgt3F, tgt3B, nullptr, M1);

    // ---- FFN ----
    launch_cvt(f1_w, wslot, (size_t)1024 * 256, stream);
    launch_gemm(tgt3B, wslot, f1_b, M1, 1024, 256, nullptr, ffnB, 1, stream);
    launch_cvt(f2_w, wslot, (size_t)256 * 1024, stream);
    launch_gemm(ffnB, wslot, f2_b, M1, 256, 1024, ffnF, nullptr, 0, stream);
    k_add_ln<<<(M1 + 7) / 8, 256, 0, stream>>>(ffnF, tgt3F, n3_g, n3_b, (float*)d_out, nullptr, nullptr, M1);
}
